// VariationalAutoEncoder_61555471286593
// MI455X (gfx1250) — compile-verified, matrix operand order checked by reading, not run on hardware
//
#include <hip/hip_runtime.h>
#include <hip/hip_bf16.h>
#include <cstdint>
#include <cstddef>

// ---------------- problem constants ----------------
#define VSZ 32000
#define ESZ 512
#define HSZ 512
#define BSZ 128
#define TSZ 23
#define G4H 2048   // 4*H

typedef __bf16 bf16_t;
typedef __bf16 v16bf __attribute__((ext_vector_type(16)));
typedef __bf16 v8bf  __attribute__((ext_vector_type(8)));
typedef float  v8f   __attribute__((ext_vector_type(8)));
typedef unsigned int v4u __attribute__((ext_vector_type(4)));
typedef int v8i __attribute__((ext_vector_type(8)));
typedef int v4i __attribute__((ext_vector_type(4)));

// LDS tile geometry: 128 rows x 32 bf16, padded to 40 halves (80B) per row so the
// 16B fragment chunks stay aligned and rows land on different bank groups.
#define TILE_HALVES (128 * 40)
#define TILE_BYTES  (TILE_HALVES * 2)

// ---------------- small elementwise kernels ----------------
__global__ __launch_bounds__(256) void k_cvt_bf16(const float* __restrict__ in,
                                                  bf16_t* __restrict__ out, int n) {
  int i = blockIdx.x * 256 + threadIdx.x;
  if (i < n) out[i] = (bf16_t)in[i];
}

__global__ __launch_bounds__(256) void k_vadd(const float* __restrict__ a,
                                              const float* __restrict__ b,
                                              float* __restrict__ o, int n) {
  int i = blockIdx.x * 256 + threadIdx.x;
  if (i < n) o[i] = a[i] + b[i];
}

__global__ __launch_bounds__(256) void k_zero_f32(float* __restrict__ p, int n) {
  int i = blockIdx.x * 256 + threadIdx.x;
  if (i < n) p[i] = 0.0f;
}

__global__ __launch_bounds__(256) void k_zero_bf16(bf16_t* __restrict__ p, int n) {
  int i = blockIdx.x * 256 + threadIdx.x;
  if (i < n) p[i] = (bf16_t)0.0f;
}

// ---------------- embedding + layernorm -> bf16 A matrix ----------------
__global__ __launch_bounds__(256) void k_embed_ln(const int* __restrict__ x,
                                                  const float* __restrict__ emb,
                                                  const float* __restrict__ gamma,
                                                  const float* __restrict__ beta,
                                                  bf16_t* __restrict__ eout) {
  __shared__ float red[256];
  const int row = blockIdx.x;            // b*T + t
  const int tid = threadIdx.x;
  const int idx = x[row];
  const float* er = emb + (size_t)idx * ESZ;
  float s = 0.f;
  for (int j = tid; j < ESZ; j += 256) s += er[j];
  red[tid] = s; __syncthreads();
  for (int st = 128; st > 0; st >>= 1) { if (tid < st) red[tid] += red[tid + st]; __syncthreads(); }
  const float mu = red[0] * (1.0f / ESZ);
  __syncthreads();
  float v = 0.f;
  for (int j = tid; j < ESZ; j += 256) { float d = er[j] - mu; v += d * d; }
  red[tid] = v; __syncthreads();
  for (int st = 128; st > 0; st >>= 1) { if (tid < st) red[tid] += red[tid + st]; __syncthreads(); }
  const float rstd = rsqrtf(red[0] * (1.0f / ESZ) + 1e-5f);
  bf16_t* orow = eout + (size_t)row * ESZ;
  for (int j = tid; j < ESZ; j += 256)
    orow[j] = (bf16_t)((er[j] - mu) * rstd * gamma[j] + beta[j]);
}

// ---------------- TDM tile load (gfx1250 Tensor Data Mover) ----------------
#if defined(__gfx1250__) && __has_builtin(__builtin_amdgcn_tensor_load_to_lds) && \
    __has_builtin(__builtin_amdgcn_s_wait_tensorcnt)
#define USE_TDM 1
#else
#define USE_TDM 0
#endif

#if USE_TDM
// DMA a 128-row x 32-element bf16 tile (row stride `ld` elements) from global
// memory into LDS at byte offset lds_off, inserting 16B of padding after every
// 64B of data (pad_interval=16 DWORDs, pad_amount=4 DWORDs) -> 80B LDS rows.
__device__ __forceinline__ void tdm_load_tile(const bf16_t* gptr, int ld, int dim1,
                                              unsigned lds_off) {
  const unsigned long long ga = (unsigned long long)(uintptr_t)gptr;
  const unsigned tdim0 = (unsigned)ld;      // dim0 extent == leading dim (K)
  const unsigned tdim1 = (unsigned)dim1;    // rows available (OOB never hit here)
  v4u g0;
  g0[0] = 1u;                                              // count=1, user mode
  g0[1] = lds_off;                                         // lds_addr
  g0[2] = (unsigned)(ga & 0xffffffffu);                    // global_addr[31:0]
  g0[3] = (unsigned)((ga >> 32) & 0x01ffffffu) | (2u << 30); // addr[56:32] | type=2
  v8i g1;
  g1[0] = (int)((1u << 16) |      // data_size = 2 bytes
                (1u << 20) |      // pad_enable
                (3u << 22) |      // pad_interval: 16 DWORDs
                (3u << 25));      // pad_amount: 4 DWORDs
  g1[1] = (int)((tdim0 & 0xffffu) << 16);                            // tensor_dim0 lo16
  g1[2] = (int)(((tdim0 >> 16) & 0xffffu) | ((tdim1 & 0xffffu) << 16)); // dim0 hi | dim1 lo
  g1[3] = (int)(((tdim1 >> 16) & 0xffffu) | (32u << 16));            // dim1 hi | tile_dim0=32
  g1[4] = (int)128u;                                                 // tile_dim1=128, tile_dim2=0
  g1[5] = (int)(unsigned)ld;                                         // tensor_dim0_stride lo32
  g1[6] = 0;                                                         // stride hi / dim1_stride lo
  g1[7] = 0;
  v4i gz4 = {0, 0, 0, 0};
  v8i gz8 = {0, 0, 0, 0, 0, 0, 0, 0};
  // 6-arg form (clang-23 / therock-10.0 headers): groups 0..3 + extra group + cpol
  __builtin_amdgcn_tensor_load_to_lds(g0, g1, gz4, gz4, gz8, 0);
}
#endif

// ---------------- generic WMMA bf16 GEMM: C[M,N] = A[M,K] * Bw[N,K]^T + bias ----------------
// Block tile 128x128, 8 waves (2x4), wave tile 64x32 (4x2 fragments), K in 32-chunks,
// double-buffered LDS tiles fed by the Tensor Data Mover (wave 0 issues, TENSORcnt gates).
__device__ __forceinline__ v16bf ldfrag(const bf16_t* p, int kh) {
  // 16-bit A/B fragment layout: lanes 0-15 take K {kh..kh+7, 16+kh..23+kh}, kh=(lane/16)*8
  v8bf lo = *(const v8bf*)(p + kh);
  v8bf hi = *(const v8bf*)(p + 16 + kh);
  return __builtin_shufflevector(lo, hi, 0,1,2,3,4,5,6,7,8,9,10,11,12,13,14,15);
}

__global__ __launch_bounds__(256) void k_gemm_bf16(const bf16_t* __restrict__ A, int lda,
                                                   const bf16_t* __restrict__ Bw, int ldb,
                                                   const float* __restrict__ bias,
                                                   float* __restrict__ C, int ldc,
                                                   int K, int M, int N) {
  __shared__ bf16_t sA[2][TILE_HALVES];
  __shared__ bf16_t sB[2][TILE_HALVES];
  const int tid  = threadIdx.x;
  const int lane = tid & 31;
  const int wid  = tid >> 5;
  const int wm   = wid >> 2;        // 0..1
  const int wn   = wid & 3;         // 0..3
  const int m0   = blockIdx.y * 128;
  const int n0   = blockIdx.x * 128;
  const int lrow = lane & 15;
  const int lkh  = (lane >> 4) << 3;   // 0 or 8
  const int nk   = K >> 5;             // K / 32

  v8f acc[4][2] = {};

#if USE_TDM
  const bf16_t* Abase = A  + (size_t)m0 * lda;
  const bf16_t* Bbase = Bw + (size_t)n0 * ldb;
  const unsigned ldsA0 = (unsigned)(uintptr_t)(__attribute__((address_space(3))) bf16_t*)&sA[0][0];
  const unsigned ldsA1 = (unsigned)(uintptr_t)(__attribute__((address_space(3))) bf16_t*)&sA[1][0];
  const unsigned ldsB0 = (unsigned)(uintptr_t)(__attribute__((address_space(3))) bf16_t*)&sB[0][0];
  const unsigned ldsB1 = (unsigned)(uintptr_t)(__attribute__((address_space(3))) bf16_t*)&sB[1][0];
  if (wid == 0) {                       // TDM issue wave (EXEC ignored by TDM)
    tdm_load_tile(Abase,      lda, M - m0, ldsA0);
    tdm_load_tile(Bbase,      ldb, N - n0, ldsB0);
    if (nk > 1) {
      tdm_load_tile(Abase + 32, lda, M - m0, ldsA1);
      tdm_load_tile(Bbase + 32, ldb, N - n0, ldsB1);
    }
  }
  for (int i = 0; i < nk; ++i) {
    if (wid == 0) {
      if (i + 1 < nk) __builtin_amdgcn_s_wait_tensorcnt((short)2);  // chunk i landed
      else            __builtin_amdgcn_s_wait_tensorcnt((short)0);
    }
    __syncthreads();
    const bf16_t* bufA = sA[i & 1];
    const bf16_t* bufB = sB[i & 1];
    v16bf af[4], bfr[2];
#pragma unroll
    for (int a = 0; a < 4; ++a) af[a]  = ldfrag(&bufA[(wm * 64 + a * 16 + lrow) * 40], lkh);
#pragma unroll
    for (int b = 0; b < 2; ++b) bfr[b] = ldfrag(&bufB[(wn * 32 + b * 16 + lrow) * 40], lkh);
#pragma unroll
    for (int a = 0; a < 4; ++a)
#pragma unroll
      for (int b = 0; b < 2; ++b)
        acc[a][b] = __builtin_amdgcn_wmma_f32_16x16x32_bf16(
            false, af[a], false, bfr[b], (short)0, acc[a][b], false, false);
    __syncthreads();
    if (wid == 0 && i + 2 < nk) {       // refill the buffer everyone just finished reading
      const int kc = (i + 2) * 32;
      tdm_load_tile(Abase + kc, lda, M - m0, (i & 1) ? ldsA1 : ldsA0);
      tdm_load_tile(Bbase + kc, ldb, N - n0, (i & 1) ? ldsB1 : ldsB0);
    }
  }
#else
  // Fallback: classic global->VGPR->LDS staging (single buffer).
  for (int kc = 0; kc < K; kc += 32) {
#pragma unroll
    for (int c = 0; c < 2; ++c) {
      const int ci  = tid + c * 256;
      const int row = ci >> 2;           // 0..127
      const int kp  = (ci & 3) << 3;     // 0,8,16,24
      *(v8bf*)&sA[0][row * 40 + kp] = *(const v8bf*)(A  + (size_t)(m0 + row) * lda + kc + kp);
      *(v8bf*)&sB[0][row * 40 + kp] = *(const v8bf*)(Bw + (size_t)(n0 + row) * ldb + kc + kp);
    }
    __syncthreads();
    v16bf af[4], bfr[2];
#pragma unroll
    for (int a = 0; a < 4; ++a) af[a]  = ldfrag(&sA[0][(wm * 64 + a * 16 + lrow) * 40], lkh);
#pragma unroll
    for (int b = 0; b < 2; ++b) bfr[b] = ldfrag(&sB[0][(wn * 32 + b * 16 + lrow) * 40], lkh);
#pragma unroll
    for (int a = 0; a < 4; ++a)
#pragma unroll
      for (int b = 0; b < 2; ++b)
        acc[a][b] = __builtin_amdgcn_wmma_f32_16x16x32_bf16(
            false, af[a], false, bfr[b], (short)0, acc[a][b], false, false);
    __syncthreads();
  }
#endif

  // writeback: VGPR r -> M = r (lanes 0-15) / r+8 (lanes 16-31), N = lane%16
#pragma unroll
  for (int i = 0; i < 4; ++i) {
    const int rbase = m0 + wm * 64 + i * 16 + ((lane >> 4) << 3);
#pragma unroll
    for (int j = 0; j < 2; ++j) {
      const int col = n0 + wn * 32 + j * 16 + lrow;
      const float bv = bias ? bias[col] : 0.0f;
#pragma unroll
      for (int r = 0; r < 8; ++r)
        C[(size_t)(rbase + r) * ldc + col] = acc[i][j][r] + bv;
    }
  }
}

// ---------------- LSTM cell elementwise ----------------
__device__ __forceinline__ float sigf(float x) { return 1.0f / (1.0f + expf(-x)); }

__global__ __launch_bounds__(256) void k_lstm_cell(const float* __restrict__ gx, int gx_stride,
                                                   const float* __restrict__ ghh,
                                                   float* __restrict__ c,
                                                   bf16_t* __restrict__ h_bf,
                                                   float* __restrict__ h_f32) {
  const int t = blockIdx.x * 256 + threadIdx.x;   // b*H + j   (B*H = 65536 threads)
  const int b = t >> 9;
  const int j = t & 511;
  const float* g0 = gx  + (size_t)b * gx_stride;
  const float* g1 = ghh + (size_t)b * G4H;
  const float ig = sigf(g0[j]            + g1[j]);
  const float fg = sigf(g0[HSZ + j]      + g1[HSZ + j]);
  const float gg = tanhf(g0[2 * HSZ + j] + g1[2 * HSZ + j]);
  const float og = sigf(g0[3 * HSZ + j]  + g1[3 * HSZ + j]);
  const float cn = fg * c[t] + ig * gg;
  c[t] = cn;
  const float h = og * tanhf(cn);
  h_bf[t] = (bf16_t)h;
  if (h_f32) h_f32[t] = h;
}

// ---------------- context gather: hs[x_lens-1, b] ----------------
__global__ __launch_bounds__(256) void k_gather_ctx(const int* __restrict__ x_lens,
                                                    const float* __restrict__ hs,
                                                    bf16_t* __restrict__ ctx) {
  const int t = blockIdx.x * 256 + threadIdx.x;   // b*H + j
  const int b = t >> 9;
  const int len = x_lens[b];
  ctx[t] = (bf16_t)hs[(size_t)(len - 1) * BSZ * HSZ + t];
}

// ---------------- reparameterization ----------------
__global__ __launch_bounds__(256) void k_reparam(const float* __restrict__ zm,
                                                 const float* __restrict__ zlv,
                                                 const float* __restrict__ eps,
                                                 float* __restrict__ enc_out,
                                                 bf16_t* __restrict__ enc_bf) {
  const int i = blockIdx.x * 256 + threadIdx.x;
  const float e = zm[i] + eps[i] * expf(0.5f * zlv[i]);
  enc_out[i] = e;
  enc_bf[i]  = (bf16_t)e;
}

// ---------------- row argmax (first-index tie-break) ----------------
__global__ __launch_bounds__(256) void k_argmax(const float* __restrict__ logits,
                                                float* __restrict__ tok) {
  __shared__ float sv[256];
  __shared__ int   si[256];
  const int row = blockIdx.x;
  const int tid = threadIdx.x;
  const float* r = logits + (size_t)row * VSZ;
  float best = -3.4e38f; int bi = VSZ;
  for (int j = tid; j < VSZ; j += 256) {
    const float v = r[j];
    if (v > best) { best = v; bi = j; }
  }
  sv[tid] = best; si[tid] = bi; __syncthreads();
  for (int st = 128; st > 0; st >>= 1) {
    if (tid < st) {
      if (sv[tid + st] > sv[tid] || (sv[tid + st] == sv[tid] && si[tid + st] < si[tid])) {
        sv[tid] = sv[tid + st]; si[tid] = si[tid + st];
      }
    }
    __syncthreads();
  }
  if (tid == 0) tok[row] = (float)si[0];
}

// ---------------- host orchestration ----------------
extern "C" void kernel_launch(void* const* d_in, const int* in_sizes, int n_in,
                              void* d_out, int out_size, void* d_ws, size_t ws_size,
                              hipStream_t stream) {
  (void)in_sizes; (void)n_in; (void)out_size; (void)ws_size;
  const int*   x        = (const int*)d_in[0];
  const int*   x_lens   = (const int*)d_in[1];
  const float* eps      = (const float*)d_in[2];
  const float* emb      = (const float*)d_in[3];
  const float* ln_gamma = (const float*)d_in[4];
  const float* ln_beta  = (const float*)d_in[5];
  const float* enc_Wih  = (const float*)d_in[6];
  const float* enc_Whh  = (const float*)d_in[7];
  const float* enc_bih  = (const float*)d_in[8];
  const float* enc_bhh  = (const float*)d_in[9];
  const float* zm_W     = (const float*)d_in[10];
  const float* zm_b     = (const float*)d_in[11];
  const float* zlv_W    = (const float*)d_in[12];
  const float* zlv_b    = (const float*)d_in[13];
  const float* dec_Wih  = (const float*)d_in[14];
  const float* dec_Whh  = (const float*)d_in[15];
  const float* dec_bih  = (const float*)d_in[16];
  const float* dec_bhh  = (const float*)d_in[17];
  const float* lin_W    = (const float*)d_in[18];
  const float* lin_b    = (const float*)d_in[19];
  // reemb_W / reemb_b (d_in[20..21]) are unused by the reference outputs.

  // d_out layout: encoded[128,512] | z_mean | z_log_var | tokens[23,128] | logits[23,128,32000]
  float* out       = (float*)d_out;
  float* o_encoded = out;
  float* o_zmean   = out + 65536;
  float* o_zlv     = out + 131072;
  float* o_tok     = out + 196608;
  float* o_logits  = out + 199552;

  // workspace bump allocator
  char* w = (char*)d_ws;
  auto alloc = [&](size_t bytes) -> char* {
    char* p = w;
    w += (bytes + 255) & ~(size_t)255;
    return p;
  };
  bf16_t* e_bf       = (bf16_t*)alloc((size_t)BSZ * TSZ * ESZ * 2);
  bf16_t* enc_Wih_bf = (bf16_t*)alloc((size_t)G4H * ESZ * 2);
  bf16_t* enc_Whh_bf = (bf16_t*)alloc((size_t)G4H * HSZ * 2);
  bf16_t* zm_W_bf    = (bf16_t*)alloc((size_t)HSZ * HSZ * 2);
  bf16_t* zlv_W_bf   = (bf16_t*)alloc((size_t)HSZ * HSZ * 2);
  bf16_t* dec_Wih_bf = (bf16_t*)alloc((size_t)G4H * HSZ * 2);
  bf16_t* dec_Whh_bf = (bf16_t*)alloc((size_t)G4H * HSZ * 2);
  bf16_t* lin_W_bf   = (bf16_t*)alloc((size_t)VSZ * HSZ * 2);
  float*  gx_enc     = (float*)alloc((size_t)BSZ * TSZ * G4H * 4);
  float*  ghh        = (float*)alloc((size_t)BSZ * G4H * 4);
  float*  c_buf      = (float*)alloc((size_t)BSZ * HSZ * 4);
  bf16_t* h_enc_bf   = (bf16_t*)alloc((size_t)BSZ * HSZ * 2);
  float*  hs_f32     = (float*)alloc((size_t)TSZ * BSZ * HSZ * 4);
  bf16_t* ctx_bf     = (bf16_t*)alloc((size_t)BSZ * HSZ * 2);
  bf16_t* enc_bf     = (bf16_t*)alloc((size_t)BSZ * HSZ * 2);
  float*  gx_dec     = (float*)alloc((size_t)BSZ * G4H * 4);
  bf16_t* hs_dec_bf  = (bf16_t*)alloc((size_t)TSZ * BSZ * HSZ * 2);
  bf16_t* zbf        = (bf16_t*)alloc((size_t)BSZ * HSZ * 2);
  float*  bsum_enc   = (float*)alloc((size_t)G4H * 4);
  float*  bsum_dec   = (float*)alloc((size_t)G4H * 4);

  const int BH = BSZ * HSZ;   // 65536

  // ---- weight conversion fp32 -> bf16 ----
  auto cvt = [&](const float* src, bf16_t* dst, int n) {
    k_cvt_bf16<<<(n + 255) / 256, 256, 0, stream>>>(src, dst, n);
  };
  cvt(enc_Wih, enc_Wih_bf, G4H * ESZ);
  cvt(enc_Whh, enc_Whh_bf, G4H * HSZ);
  cvt(zm_W,    zm_W_bf,    HSZ * HSZ);
  cvt(zlv_W,   zlv_W_bf,   HSZ * HSZ);
  cvt(dec_Wih, dec_Wih_bf, G4H * HSZ);
  cvt(dec_Whh, dec_Whh_bf, G4H * HSZ);
  cvt(lin_W,   lin_W_bf,   VSZ * HSZ);

  k_vadd<<<(G4H + 255) / 256, 256, 0, stream>>>(enc_bih, enc_bhh, bsum_enc, G4H);
  k_vadd<<<(G4H + 255) / 256, 256, 0, stream>>>(dec_bih, dec_bhh, bsum_dec, G4H);
  k_zero_bf16<<<BH / 256, 256, 0, stream>>>(zbf, BH);
  k_zero_f32<<<BH / 256, 256, 0, stream>>>(c_buf, BH);

  // ---- embedding + layernorm ----
  k_embed_ln<<<BSZ * TSZ, 256, 0, stream>>>(x, emb, ln_gamma, ln_beta, e_bf);

  // ---- gx_enc = e @ enc_Wih^T + (bih + bhh)  : [2944,2048] ----
  k_gemm_bf16<<<dim3(G4H / 128, (BSZ * TSZ) / 128), 256, 0, stream>>>(
      e_bf, ESZ, enc_Wih_bf, ESZ, bsum_enc, gx_enc, G4H, ESZ, BSZ * TSZ, G4H);

  // ---- encoder scan: 23 steps ----
  for (int t = 0; t < TSZ; ++t) {
    const bf16_t* hin = (t == 0) ? zbf : h_enc_bf;
    k_gemm_bf16<<<dim3(G4H / 128, 1), 256, 0, stream>>>(
        hin, HSZ, enc_Whh_bf, HSZ, nullptr, ghh, G4H, HSZ, BSZ, G4H);
    k_lstm_cell<<<BH / 256, 256, 0, stream>>>(
        gx_enc + (size_t)t * G4H, TSZ * G4H, ghh, c_buf, h_enc_bf, hs_f32 + (size_t)t * BH);
  }

  // ---- context gather + VAE code layer ----
  k_gather_ctx<<<BH / 256, 256, 0, stream>>>(x_lens, hs_f32, ctx_bf);
  k_gemm_bf16<<<dim3(HSZ / 128, 1), 256, 0, stream>>>(
      ctx_bf, HSZ, zm_W_bf, HSZ, zm_b, o_zmean, HSZ, HSZ, BSZ, HSZ);
  k_gemm_bf16<<<dim3(HSZ / 128, 1), 256, 0, stream>>>(
      ctx_bf, HSZ, zlv_W_bf, HSZ, zlv_b, o_zlv, HSZ, HSZ, BSZ, HSZ);
  k_reparam<<<BH / 256, 256, 0, stream>>>(o_zmean, o_zlv, eps, o_encoded, enc_bf);

  // ---- gx_dec = encoded @ dec_Wih^T + (bih + bhh) : [128,2048] ----
  k_gemm_bf16<<<dim3(G4H / 128, 1), 256, 0, stream>>>(
      enc_bf, HSZ, dec_Wih_bf, HSZ, bsum_dec, gx_dec, G4H, HSZ, BSZ, G4H);

  // ---- decoder scan: 23 steps ----
  k_zero_f32<<<BH / 256, 256, 0, stream>>>(c_buf, BH);
  for (int t = 0; t < TSZ; ++t) {
    const bf16_t* hin = (t == 0) ? zbf : (hs_dec_bf + (size_t)(t - 1) * BH);
    k_gemm_bf16<<<dim3(G4H / 128, 1), 256, 0, stream>>>(
        hin, HSZ, dec_Whh_bf, HSZ, nullptr, ghh, G4H, HSZ, BSZ, G4H);
    k_lstm_cell<<<BH / 256, 256, 0, stream>>>(
        gx_dec, G4H, ghh, c_buf, hs_dec_bf + (size_t)t * BH, nullptr);
  }

  // ---- logits = hs_dec @ lin_W^T + lin_b : [2944, 32000]  (dominant GEMM) ----
  k_gemm_bf16<<<dim3(VSZ / 128, (TSZ * BSZ) / 128), 256, 0, stream>>>(
      hs_dec_bf, HSZ, lin_W_bf, HSZ, lin_b, o_logits, VSZ, HSZ, TSZ * BSZ, VSZ);

  // ---- argmax over vocab ----
  k_argmax<<<TSZ * BSZ, 256, 0, stream>>>(o_logits, o_tok);
}
